// TemporalGraphNeuralNetwork_43121471652512
// MI455X (gfx1250) — matrix operand order repaired and checked by reading, not compiled, not run on hardware
//
#include <hip/hip_runtime.h>
#include <hip/hip_bf16.h>

// ---------------------------------------------------------------------------
// Temporal GNN for MI455X (gfx1250, wave32, WMMA).
// GEMMs: v_wmma_f32_16x16x32_bf16 with swizzled (fragment-ready) bf16 weights
// so every B fragment is a single contiguous 32B load. Scatter is HBM-bound.
// ---------------------------------------------------------------------------

typedef __bf16 bf16_t;
typedef bf16_t v16bf __attribute__((ext_vector_type(16)));
typedef float  v8f   __attribute__((ext_vector_type(8)));

#define HD   256
#define GG   512
#define LN_EPS 1e-5f
#define LOG10000 9.210340371976184f

// --------- weight f32 -> bf16, swizzled into WMMA B-fragment order ---------
// Packed layout: out[(((nTile*8 + kb)*32 + lane)*16 + e]
//   lane = g*16 + lrow ; n = nTile*16 + lrow ; k = kb*32 + g*16 + e
// so a lane's 16 B-fragment elements are contiguous (one aligned 32B load).
__global__ void pack_w_kernel(const float* __restrict__ W, bf16_t* __restrict__ out) {
    int idx = blockIdx.x * blockDim.x + threadIdx.x;     // 0..4095
    if (idx >= 16 * 8 * 32) return;
    int lane  = idx & 31;
    int kb    = (idx >> 5) & 7;
    int nTile = idx >> 8;
    int g = lane >> 4, lrow = lane & 15;
    int n = nTile * 16 + lrow;
    bf16_t tmp[16];
    #pragma unroll
    for (int e = 0; e < 16; ++e) {
        int k = kb * 32 + g * 16 + e;
        tmp[e] = (bf16_t)W[k * HD + n];
    }
    bf16_t* dst = out + (size_t)idx * 16;
    #pragma unroll
    for (int e = 0; e < 16; ++e) dst[e] = tmp[e];
}

// ------------------------- degree / dinv -----------------------------------
__global__ void deg_kernel(const int* __restrict__ dst, float* __restrict__ deg, int E) {
    int e = blockIdx.x * blockDim.x + threadIdx.x;
    if (e < E) atomicAdd(&deg[dst[e]], 1.0f);
}
__global__ void dinv_kernel(float* __restrict__ degdinv, int n) {
    int i = blockIdx.x * blockDim.x + threadIdx.x;
    if (i < n) degdinv[i] = __frsqrt_rn(degdinv[i] + 1.0f);  // +1 self-loop => deg>=1
}

// ---- shared A-fragment loader (contiguous k = 16g + e, same order as packed B) ----
__device__ __forceinline__ v16bf load_a_frag(const float* __restrict__ src,
                                             int arow, int kBase, int g) {
    v16bf a;
    #pragma unroll
    for (int e = 0; e < 16; ++e) {
        // lanes0-15: K = {0..15}; lanes16-31: K = {16..31}
        int k = kBase + g * 16 + e;
        a[e] = (bf16_t)src[(size_t)arow * HD + k];
    }
    return a;
}

// ------------- feature transform: GEMM + bias + ReLU + LN + PE -------------
// grid.x = ceil(N/16), 256 threads (8 waves). Wave w: cols [w*32, w*32+32).
__global__ __launch_bounds__(256) void ft_kernel(
    const float* __restrict__ x, const bf16_t* __restrict__ Wp,
    const float* __restrict__ bias, const float* __restrict__ ln_g,
    const float* __restrict__ ln_b, float* __restrict__ hout, int n)
{
    __shared__ float sh[16 * HD];
    __shared__ float redS[256], redQ[256];
    __shared__ float smu[16], srs[16];
    const int tid  = threadIdx.x;
    const int wave = tid >> 5, lane = tid & 31;
    const int g    = lane >> 4, lrow = lane & 15;
    const int r0   = blockIdx.x * 16;
    const int n0   = wave * 32;
    const int nT0  = wave * 2;

    int arow = r0 + lrow; if (arow >= n) arow = n - 1;   // clamp: EXEC all-1 for WMMA

    const v16bf* Bp = (const v16bf*)Wp;
    v8f c0 = {}; v8f c1 = {};
    for (int kb = 0; kb < 8; ++kb) {
        v16bf a  = load_a_frag(x, arow, kb * 32, g);
        v16bf b0 = Bp[((nT0       * 8 + kb) * 32) + lane];   // one 32B contiguous load
        v16bf b1 = Bp[(((nT0 + 1) * 8 + kb) * 32) + lane];
        c0 = __builtin_amdgcn_wmma_f32_16x16x32_bf16(false, a, false, b0, (short)0, c0, false, false);
        c1 = __builtin_amdgcn_wmma_f32_16x16x32_bf16(false, a, false, b1, (short)0, c1, false, false);
    }
    // C/D layout: lane=N, VGPR v -> M=v (lanes 0-15) or M=8+v (lanes 16-31)
    #pragma unroll
    for (int v = 0; v < 8; ++v) {
        int m = (g == 0) ? v : (8 + v);
        sh[m * HD + n0 + lrow]      = c0[v];
        sh[m * HD + n0 + 16 + lrow] = c1[v];
    }
    __syncthreads();
    {   // bias + ReLU, thread owns column `tid`
        float bb = bias[tid];
        #pragma unroll
        for (int m = 0; m < 16; ++m) {
            float v = sh[m * HD + tid] + bb;
            sh[m * HD + tid] = v > 0.f ? v : 0.f;
        }
    }
    __syncthreads();
    {   // LN stats: 16 threads per row, 16-element partials, then reduce
        int m = tid >> 4, seg = tid & 15;
        float s = 0.f, ss = 0.f;
        #pragma unroll
        for (int c = seg * 16; c < seg * 16 + 16; ++c) {
            float v = sh[m * HD + c]; s += v; ss += v * v;
        }
        redS[tid] = s; redQ[tid] = ss;
    }
    __syncthreads();
    if (tid < 16) {
        float s = 0.f, ss = 0.f;
        #pragma unroll
        for (int j = 0; j < 16; ++j) { s += redS[tid * 16 + j]; ss += redQ[tid * 16 + j]; }
        float mu  = s * (1.f / HD);
        float var = ss * (1.f / HD) - mu * mu;
        smu[tid] = mu;
        srs[tid] = __frsqrt_rn(var + LN_EPS);
    }
    __syncthreads();
    {   // normalize + gamma/beta + sinusoidal positional encoding
        float gga = ln_g[tid], bbe = ln_b[tid];
        float j   = (float)(tid & ~1);
        float div = __expf(j * (-LOG10000 / (float)HD));
        for (int m = 0; m < 16; ++m) {
            int node = r0 + m;
            if (node >= n) break;
            float ang = (float)node * div;
            float pe  = (tid & 1) ? __cosf(ang) : __sinf(ang);
            float v   = (sh[m * HD + tid] - smu[m]) * srs[m] * gga + bbe + pe;
            hout[(size_t)node * HD + tid] = v;
        }
    }
}

// ---------------- GCN GEMM: t = (h @ W)*dinv[row]; acc = t*dinv[row] -------
__global__ __launch_bounds__(256) void gcn_gemm_kernel(
    const float* __restrict__ hin, const bf16_t* __restrict__ Wp,
    const float* __restrict__ dinv, float* __restrict__ tout,
    float* __restrict__ acc, int n)
{
    const int tid  = threadIdx.x;
    const int wave = tid >> 5, lane = tid & 31;
    const int g    = lane >> 4, lrow = lane & 15;
    const int r0   = blockIdx.x * 16;
    const int n0   = wave * 32;
    const int nT0  = wave * 2;

    int arow = r0 + lrow; if (arow >= n) arow = n - 1;

    const v16bf* Bp = (const v16bf*)Wp;
    v8f c0 = {}; v8f c1 = {};
    for (int kb = 0; kb < 8; ++kb) {
        v16bf a  = load_a_frag(hin, arow, kb * 32, g);
        v16bf b0 = Bp[((nT0       * 8 + kb) * 32) + lane];
        v16bf b1 = Bp[(((nT0 + 1) * 8 + kb) * 32) + lane];
        c0 = __builtin_amdgcn_wmma_f32_16x16x32_bf16(false, a, false, b0, (short)0, c0, false, false);
        c1 = __builtin_amdgcn_wmma_f32_16x16x32_bf16(false, a, false, b1, (short)0, c1, false, false);
    }
    #pragma unroll
    for (int v = 0; v < 8; ++v) {
        int m = (g == 0) ? v : (8 + v);
        int node = r0 + m;
        if (node >= n) continue;
        float dv = dinv[node];
        size_t o0 = (size_t)node * HD + n0 + lrow;
        float t0 = c0[v] * dv, t1 = c1[v] * dv;
        tout[o0]      = t0;        tout[o0 + 16] = t1;        // message pre-scaled by dinv[src]
        acc[o0]       = t0 * dv;   acc[o0 + 16]  = t1 * dv;   // self-loop: dinv[i]^2 * (hW)[i]
    }
}

// ---------------- edge scatter: acc[dst] += t[src] * dinv[dst] -------------
// 4 edges per 256-thread block, 64 lanes/edge, float4 per lane.
__global__ __launch_bounds__(256) void scatter_kernel(
    const float* __restrict__ t, const float* __restrict__ dinv,
    const int* __restrict__ src, const int* __restrict__ dst,
    float* __restrict__ acc, int E)
{
    int tid = threadIdx.x;
    int e = blockIdx.x * 4 + (tid >> 6);
    if (e >= E) return;
    int s = src[e], d = dst[e];
    float sc = dinv[d];
    int c0 = (tid & 63) * 4;
    const float4 v = *(const float4*)(t + (size_t)s * HD + c0);
    float* a = acc + (size_t)d * HD + c0;
    atomicAdd(a + 0, v.x * sc);
    atomicAdd(a + 1, v.y * sc);
    atomicAdd(a + 2, v.z * sc);
    atomicAdd(a + 3, v.w * sc);
}

// ---------------- h = relu(acc + bias) -------------------------------------
__global__ void bias_relu_kernel(const float* __restrict__ acc, const float* __restrict__ bias,
                                 float* __restrict__ hout, int total) {
    int i = blockIdx.x * blockDim.x + threadIdx.x;
    if (i >= total) return;
    float v = acc[i] + bias[i & (HD - 1)];
    hout[i] = v > 0.f ? v : 0.f;
}

// ---------------- global mean pool -----------------------------------------
__global__ void pool_kernel(const float* __restrict__ h, const int* __restrict__ batch,
                            float* __restrict__ pooled, float* __restrict__ cnts, int n) {
    int node = blockIdx.x;
    if (node >= n) return;
    int gi = batch[node];
    int c0 = threadIdx.x * 4;
    const float4 v = *(const float4*)(h + (size_t)node * HD + c0);
    float* p = pooled + (size_t)gi * HD + c0;
    atomicAdd(p + 0, v.x); atomicAdd(p + 1, v.y);
    atomicAdd(p + 2, v.z); atomicAdd(p + 3, v.w);
    if (threadIdx.x == 0) atomicAdd(&cnts[gi], 1.0f);
}
__global__ void pool_div_kernel(float* __restrict__ pooled, const float* __restrict__ cnts) {
    int g = blockIdx.x, c = threadIdx.x;
    float cnt = cnts[g]; cnt = cnt > 1.f ? cnt : 1.f;
    pooled[g * HD + c] /= cnt;
}

// ---------------- classifier MLP (tiny: 512 rows) --------------------------
__global__ void fc_kernel(const float* __restrict__ in, const float* __restrict__ W,
                          const float* __restrict__ bias, float* __restrict__ out,
                          int K, int Nout, int do_relu) {
    int gi = blockIdx.x, nc = threadIdx.x;
    float s = bias[nc];
    for (int k = 0; k < K; ++k) s += in[gi * K + k] * W[k * Nout + nc];
    if (do_relu) s = s > 0.f ? s : 0.f;
    out[gi * Nout + nc] = s;
}

// ---------------------------------------------------------------------------
extern "C" void kernel_launch(void* const* d_in, const int* in_sizes, int n_in,
                              void* d_out, int out_size, void* d_ws, size_t ws_size,
                              hipStream_t stream) {
    const float* x    = (const float*)d_in[0];
    const float* W_ft = (const float*)d_in[1];
    const float* b_ft = (const float*)d_in[2];
    const float* ln_g = (const float*)d_in[3];
    const float* ln_b = (const float*)d_in[4];
    const float* W_g[3]  = { (const float*)d_in[5], (const float*)d_in[7], (const float*)d_in[9] };
    const float* b_g[3]  = { (const float*)d_in[6], (const float*)d_in[8], (const float*)d_in[10] };
    const float* W_c0 = (const float*)d_in[11];
    const float* b_c0 = (const float*)d_in[12];
    const float* W_c1 = (const float*)d_in[13];
    const float* b_c1 = (const float*)d_in[14];
    const float* W_c2 = (const float*)d_in[15];
    const float* b_c2 = (const float*)d_in[16];
    const int*   edge = (const int*)d_in[17];
    const int*   batch = (const int*)d_in[18];

    const int N = in_sizes[0] / HD;        // 100000
    const int E = in_sizes[17] / 2;        // 1600000
    const int* src = edge;
    const int* dst = edge + E;

    // ---- workspace layout (~310 MB) ----
    char* ws = (char*)d_ws;
    const size_t SZ_NH = (size_t)N * HD * sizeof(float);
    float*  h      = (float*)(ws);
    float*  t      = (float*)(ws + SZ_NH);
    float*  acc    = (float*)(ws + 2 * SZ_NH);
    float*  dinv   = (float*)(ws + 3 * SZ_NH);                 // N floats
    char*   wbase  = ws + 3 * SZ_NH + (size_t)N * sizeof(float);  // 32B-aligned
    bf16_t* wbf_ft = (bf16_t*)(wbase);
    bf16_t* wbf_g[3] = { (bf16_t*)(wbase + 1 * 131072),
                         (bf16_t*)(wbase + 2 * 131072),
                         (bf16_t*)(wbase + 3 * 131072) };
    char*   pbase  = wbase + 4 * 131072;
    float*  pooled = (float*)(pbase);                          // G*HD
    float*  cnts   = (float*)(pbase + (size_t)GG * HD * sizeof(float));
    float*  z1     = (float*)(pbase + (size_t)GG * HD * sizeof(float) + GG * sizeof(float));
    float*  z2     = (float*)((char*)z1 + (size_t)GG * HD * sizeof(float));

    // ---- weights -> bf16, swizzled into B-fragment order (L2-resident) ----
    pack_w_kernel<<<16, 256, 0, stream>>>(W_ft, wbf_ft);
    for (int L = 0; L < 3; ++L)
        pack_w_kernel<<<16, 256, 0, stream>>>(W_g[L], wbf_g[L]);

    // ---- symmetric degree normalization (self-loops folded in) ----
    hipMemsetAsync(dinv, 0, (size_t)N * sizeof(float), stream);
    deg_kernel<<<(E + 255) / 256, 256, 0, stream>>>(dst, dinv, E);
    dinv_kernel<<<(N + 255) / 256, 256, 0, stream>>>(dinv, N);

    // ---- feature transform (WMMA) + LN + PE ----
    ft_kernel<<<(N + 15) / 16, 256, 0, stream>>>(x, wbf_ft, b_ft, ln_g, ln_b, h, N);

    // ---- 3 GCN layers: WMMA GEMM -> edge scatter -> bias+ReLU ----
    for (int L = 0; L < 3; ++L) {
        gcn_gemm_kernel<<<(N + 15) / 16, 256, 0, stream>>>(h, wbf_g[L], dinv, t, acc, N);
        scatter_kernel<<<(E + 3) / 4, 256, 0, stream>>>(t, dinv, src, dst, acc, E);
        bias_relu_kernel<<<(N * HD + 255) / 256, 256, 0, stream>>>(acc, b_g[L], h, N * HD);
    }

    // ---- global mean pool ----
    hipMemsetAsync(pooled, 0, (size_t)GG * HD * sizeof(float) + GG * sizeof(float), stream);
    pool_kernel<<<N, 64, 0, stream>>>(h, batch, pooled, cnts, N);
    pool_div_kernel<<<GG, HD, 0, stream>>>(pooled, cnts);

    // ---- classifier MLP ----
    fc_kernel<<<GG, HD, 0, stream>>>(pooled, W_c0, b_c0, z1, HD, HD, 1);
    fc_kernel<<<GG, HD / 2, 0, stream>>>(z1, W_c1, b_c1, z2, HD, HD / 2, 1);
    fc_kernel<<<GG, 4, 0, stream>>>(z2, W_c2, b_c2, (float*)d_out, HD / 2, 4, 0);
}
